// Graph_Learn1_57664230916911
// MI455X (gfx1250) — matrix operand order repaired and checked by reading, not compiled, not run on hardware
//
#include <hip/hip_runtime.h>
#include <hip/hip_bf16.h>

typedef __attribute__((ext_vector_type(16))) __bf16          v16bf;
typedef __attribute__((ext_vector_type(8)))  __bf16          v8bf;
typedef __attribute__((ext_vector_type(8)))  float           v8f;
typedef __attribute__((ext_vector_type(4)))  unsigned short  us4;

#define NB 64
#define NV 512
#define NF 512
// x0[b,i,f] = x[b*2*NV*NF + i*NF + f]   (t = 0 slice of (B,2,V,F))

// ---------------------------------------------------------------------------
// Kernel 1: rinv[i] = rsqrt( sum_{b,f} x0[b,i,f]^2 )   (512 blocks x 256 thr)
// ---------------------------------------------------------------------------
__global__ __launch_bounds__(256)
void norm_rinv_kernel(const float* __restrict__ x, float* __restrict__ rinv) {
  const int i = blockIdx.x;
  const size_t rowbase = (size_t)i * NF;
  float s = 0.0f;
  for (int e = threadIdx.x; e < NB * NF; e += 256) {
    const int b = e >> 9;          // / NF
    const int f = e & (NF - 1);    // consecutive tid -> consecutive f (coalesced)
    const float v = x[(size_t)b * (2u * NV * NF) + rowbase + f];
    s += v * v;
  }
  __shared__ float red[256];
  red[threadIdx.x] = s;
  __syncthreads();
  for (int off = 128; off > 0; off >>= 1) {
    if (threadIdx.x < off) red[threadIdx.x] += red[threadIdx.x + off];
    __syncthreads();
  }
  if (threadIdx.x == 0) rinv[i] = rsqrtf(red[0]);
}

// ---------------------------------------------------------------------------
// bf16 hi/lo split: x ~= hi + lo, hi = trunc-bf16(x), lo = trunc-bf16(x - hi)
// ---------------------------------------------------------------------------
struct HL { unsigned short h, l; };

__device__ inline HL split_bf16(float v) {
  HL r;
  const unsigned int u = __float_as_uint(v);
  r.h = (unsigned short)(u >> 16);
  const float hf = __uint_as_float((unsigned int)r.h << 16);
  const float rem = v - hf;
  r.l = (unsigned short)(__float_as_uint(rem) >> 16);
  return r;
}

// LDS tile geometry: 128 rows x 32 cols (bf16), row stride 40 shorts (80 B).
// 80 B stride => 16-B aligned b128 reads, and (20*L mod 64) banks all distinct.
#define LDSTRIDE 40
#define TILEBYTES (128 * LDSTRIDE * 2)   // 10240 B per bf16 tile

// A-operand fragment (ISA 7.12.2, 16-bit A 16x32):
//   lanes 0-15  hold row M=lane,    K = 0..7  (v[0..7])  and 16..23 (v[8..15])
//   lanes 16-31 hold row M=lane-16, K = 8..15 (v[0..7])  and 24..31 (v[8..15])
__device__ inline v16bf ld_frag_a(const unsigned short* tile, int row, int half) {
  const __bf16* p = (const __bf16*)(tile + row * LDSTRIDE);
  const v8bf x0 = *(const v8bf*)(p + half * 8);
  const v8bf x1 = *(const v8bf*)(p + 16 + half * 8);
  v16bf f;
#pragma unroll
  for (int i = 0; i < 8; ++i) { f[i] = x0[i]; f[i + 8] = x1[i]; }
  return f;
}

// B-operand fragment (32x16, column n = lane%16 = our row j):
//   lanes 0-15 hold K = 0..7, 16..23; lanes 16-31 hold K = 8..15, 24..31 (matches A)
__device__ inline v16bf ld_frag_b(const unsigned short* tile, int row, int half) {
  const __bf16* p = (const __bf16*)(tile + row * LDSTRIDE);
  const v8bf x0 = *(const v8bf*)(p + half * 8);
  const v8bf x1 = *(const v8bf*)(p + 16 + half * 8);
  v16bf f;
#pragma unroll
  for (int i = 0; i < 8; ++i) { f[i] = x0[i]; f[i + 8] = x1[i]; }
  return f;
}

// ---------------------------------------------------------------------------
// Kernel 2: per batch b, C = x0[b] * x0[b]^T scaled by rinv[i]*rinv[j].
// Symmetry: only upper-triangle 128x128 tiles (10 of 16), mirror the rest.
// Grid (10, 64); WG = 256 thr = 8 waves; WG tile 128x128; wave tile 64x32.
// bf16x3: acc += Ahi*Bhi + Ahi*Blo + Alo*Bhi  (f32 accumulate)
// ---------------------------------------------------------------------------
__global__ __launch_bounds__(256)
void gram_bf16x3_kernel(const float* __restrict__ x,
                        const float* __restrict__ rinv,
                        float* __restrict__ out) {
  __shared__ __align__(16) unsigned char smem[4 * TILEBYTES];  // 40 KB
  unsigned short* Ahi = (unsigned short*)(smem + 0 * TILEBYTES);
  unsigned short* Alo = (unsigned short*)(smem + 1 * TILEBYTES);
  unsigned short* Bhi = (unsigned short*)(smem + 2 * TILEBYTES);
  unsigned short* Blo = (unsigned short*)(smem + 3 * TILEBYTES);

  // Map linear upper-triangle tile id -> (ti, tj), tj >= ti, 4x4 tile grid.
  int rem = blockIdx.x, ti = 0;
  while (rem >= (4 - ti)) { rem -= (4 - ti); ++ti; }
  const int tj = ti + rem;
  const int b  = blockIdx.y;

  const size_t xb    = (size_t)b * (2u * NV * NF);
  const size_t abase = xb + (size_t)(ti * 128) * NF;
  const size_t bbase = xb + (size_t)(tj * 128) * NF;

  const int tid  = threadIdx.x;
  const int lane = tid & 31;
  const int w    = tid >> 5;          // wave 0..7
  const int wr   = w >> 2;            // 0..1 : wave row (64 rows each)
  const int wc   = w & 3;             // 0..3 : wave col (32 cols each)
  const int half = lane >> 4;
  const int lr   = lane & 15;

  v8f acc[4][2];
  const v8f vzero = {0.f, 0.f, 0.f, 0.f, 0.f, 0.f, 0.f, 0.f};
#pragma unroll
  for (int t = 0; t < 4; ++t)
#pragma unroll
    for (int u = 0; u < 2; ++u) acc[t][u] = vzero;

  for (int k0 = 0; k0 < NF; k0 += 32) {
    __syncthreads();
    // Cooperative load of 128x32 f32 A and B panels, split into hi/lo bf16.
#pragma unroll
    for (int s = 0; s < 4; ++s) {
      const int q  = tid + 256 * s;   // 0..1023 float4 slots
      const int r  = q >> 3;          // row 0..127
      const int c4 = q & 7;           // float4 within the 32-wide panel
      const float4 va = *(const float4*)(x + abase + (size_t)r * NF + k0 + c4 * 4);
      const float4 vb = *(const float4*)(x + bbase + (size_t)r * NF + k0 + c4 * 4);
      us4 ah, al, bh, bl;
      HL s0, s1, s2, s3;
      s0 = split_bf16(va.x); s1 = split_bf16(va.y);
      s2 = split_bf16(va.z); s3 = split_bf16(va.w);
      ah[0] = s0.h; al[0] = s0.l;  ah[1] = s1.h; al[1] = s1.l;
      ah[2] = s2.h; al[2] = s2.l;  ah[3] = s3.h; al[3] = s3.l;
      s0 = split_bf16(vb.x); s1 = split_bf16(vb.y);
      s2 = split_bf16(vb.z); s3 = split_bf16(vb.w);
      bh[0] = s0.h; bl[0] = s0.l;  bh[1] = s1.h; bl[1] = s1.l;
      bh[2] = s2.h; bl[2] = s2.l;  bh[3] = s3.h; bl[3] = s3.l;
      const int o = r * LDSTRIDE + c4 * 4;   // 8-byte aligned -> ds_store_b64
      *(us4*)&Ahi[o] = ah;  *(us4*)&Alo[o] = al;
      *(us4*)&Bhi[o] = bh;  *(us4*)&Blo[o] = bl;
    }
    __syncthreads();

    v16bf aH[4], aL[4], bH[2], bL[2];
#pragma unroll
    for (int t = 0; t < 4; ++t) {
      const int ar = wr * 64 + t * 16 + lr;
      aH[t] = ld_frag_a(Ahi, ar, half);
      aL[t] = ld_frag_a(Alo, ar, half);
    }
#pragma unroll
    for (int u = 0; u < 2; ++u) {
      const int br = wc * 32 + u * 16 + lr;
      bH[u] = ld_frag_b(Bhi, br, half);
      bL[u] = ld_frag_b(Blo, br, half);
    }

#pragma unroll
    for (int t = 0; t < 4; ++t) {
#pragma unroll
      for (int u = 0; u < 2; ++u) {
        acc[t][u] = __builtin_amdgcn_wmma_f32_16x16x32_bf16(
            false, aH[t], false, bH[u], (short)0, acc[t][u], false, false);
        acc[t][u] = __builtin_amdgcn_wmma_f32_16x16x32_bf16(
            false, aH[t], false, bL[u], (short)0, acc[t][u], false, false);
        acc[t][u] = __builtin_amdgcn_wmma_f32_16x16x32_bf16(
            false, aL[t], false, bH[u], (short)0, acc[t][u], false, false);
      }
    }
  }

  // ---- Epilogue. C/D layout: VGPR v holds M = half*8 + v, N = lane%16. ----
  __syncthreads();   // all waves done reading the bf16 tiles; LDS reusable

  const size_t obase = (size_t)b * NV * NV;
  const int ib0 = ti * 128 + wr * 64;    // wave's global row base (i)
  const int jb0 = tj * 128 + wc * 32;    // wave's global col base (j)

  // Scale in place and write the (i, j) orientation (coalesced).
#pragma unroll
  for (int t = 0; t < 4; ++t) {
    const int ibase = ib0 + t * 16 + half * 8;
    float ri[8];
#pragma unroll
    for (int v = 0; v < 8; ++v) ri[v] = rinv[ibase + v];
#pragma unroll
    for (int u = 0; u < 2; ++u) {
      const int j  = jb0 + u * 16 + lr;
      const float rj = rinv[j];
#pragma unroll
      for (int v = 0; v < 8; ++v) {
        const float sv = acc[t][u][v] * ri[v] * rj;
        acc[t][u][v] = sv;
        out[obase + (size_t)(ibase + v) * NV + j] = sv;
      }
    }
  }

  // Mirror: write the (j, i) orientation for off-diagonal tiles.
  // Per-wave 16x17 f32 LDS scratch (pad 17 -> conflict-free both directions).
  if (ti != tj) {
    float* scr = (float*)smem + (size_t)w * 288;   // 272 used, 288 stride
#pragma unroll
    for (int t = 0; t < 4; ++t) {
      const int ib_t = ib0 + t * 16;               // col base of mirrored tile
#pragma unroll
      for (int u = 0; u < 2; ++u) {
        const int jb_u = jb0 + u * 16;             // row base of mirrored tile
        // store transposed: element (m = half*8+v, n = lr) -> scr[n][m]
#pragma unroll
        for (int v = 0; v < 8; ++v)
          scr[lr * 17 + half * 8 + v] = acc[t][u][v];
        // same-wave LDS ops are in-order; block compiler reordering only
        asm volatile("" ::: "memory");
        // read rows coalesced: lane covers (row = half*8+v, col = lr)
#pragma unroll
        for (int v = 0; v < 8; ++v) {
          const int jr = jb_u + half * 8 + v;
          out[obase + (size_t)jr * NV + ib_t + lr] = scr[(half * 8 + v) * 17 + lr];
        }
        asm volatile("" ::: "memory");
      }
    }
  }
}

extern "C" void kernel_launch(void* const* d_in, const int* in_sizes, int n_in,
                              void* d_out, int out_size, void* d_ws, size_t ws_size,
                              hipStream_t stream) {
  (void)in_sizes; (void)n_in; (void)out_size; (void)ws_size;
  const float* x   = (const float*)d_in[0];
  float* out       = (float*)d_out;
  float* rinv      = (float*)d_ws;   // 512 floats of scratch

  hipLaunchKernelGGL(norm_rinv_kernel, dim3(NV), dim3(256), 0, stream, x, rinv);
  hipLaunchKernelGGL(gram_bf16x3_kernel, dim3(10, NB), dim3(256), 0, stream,
                     x, rinv, out);
}
